// TransformerAutoencoder_7696581394490
// MI455X (gfx1250) — matrix operand order repaired and checked by reading, not compiled, not run on hardware
//
#include <hip/hip_runtime.h>

// ---------------------------------------------------------------------------
// MI455X (gfx1250) transformer encoder forward.
// GEMMs + attention on v_wmma_f32_16x16x32_f16 (f16 in, f32 accum).
// Tile staging HBM->LDS uses CDNA5 GLOBAL_LOAD_ASYNC_TO_LDS_B128 (ASYNCcnt).
// Shapes: B=4, S=2048, H=256, NHEAD=4 (dh=64), FFN=1024, L=3, window +/-32.
// ---------------------------------------------------------------------------

typedef __attribute__((ext_vector_type(16))) _Float16 v16h;
typedef __attribute__((ext_vector_type(8)))  float    v8f;

#define WMMA_F16(A, B, C) \
    __builtin_amdgcn_wmma_f32_16x16x32_f16(false, (A), false, (B), (short)0, (C), false, false)

#define FLAG_RELU     1
#define FLAG_RESIDUAL 2
#define FLAG_PE       4

#define TOKENS 8192   // B*S
#define SEQ    2048
#define HDIM   256

// CDNA5 async copy: 16B global -> LDS per lane, tracked by ASYNCcnt.
// (low 32 bits of a generic pointer to __shared__ == wave-relative LDS offset)
__device__ __forceinline__ void async_copy16(unsigned ldsOff, const float* gptr) {
    asm volatile("global_load_async_to_lds_b128 %0, %1, off"
                 :: "v"(ldsOff), "v"(gptr) : "memory");
}
__device__ __forceinline__ void wait_async() {
    asm volatile("s_wait_asynccnt 0x0" ::: "memory");
}

// ---------------------------------------------------------------------------
// Fused GEMM: out[M,N] = epilogue( A[M,K] @ W[N,K]^T + bias[N] )
// Block tile 128x64, 8 waves; wave w computes rows [16w,16w+16) x all 64 cols
// (one A fragment feeds 4 WMMAs). K staged in 32-chunks via async-to-LDS.
// FLAGS is a compile-time epilogue selector (no runtime branches).
// ---------------------------------------------------------------------------
template <int FLAGS>
__global__ __launch_bounds__(256)
void gemm_wmma(const float* __restrict__ A, const float* __restrict__ W,
               const float* __restrict__ bias, const float* __restrict__ residual,
               float* __restrict__ out, int M, int N, int K)
{
    constexpr int LDA = 40;                       // fp32 row stride (160B, 16B-aligned)
    __shared__ __align__(16) float As[128 * LDA]; // 20 KB
    __shared__ __align__(16) float Ws[64 * LDA];  // 10 KB

    const int tid   = threadIdx.x;
    const int wave  = tid >> 5;        // 0..7 -> M sub-tile
    const int lane  = tid & 31;
    const int lhalf = lane >> 4;
    const int lmod  = lane & 15;
    const int m0    = blockIdx.x * 128;
    const int n0    = blockIdx.y * 64;

    v8f acc[4] = {{}, {}, {}, {}};

    for (int k0 = 0; k0 < K; k0 += 32) {
        // ---- async-stage A (128x32) and W (64x32) fp32 tiles into LDS.
        // 16B chunks: A = 1024 chunks (4/thread), W = 512 chunks (2/thread).
#pragma unroll
        for (int i = 0; i < 4; ++i) {
            const int id = tid + 256 * i, row = id >> 3, q = id & 7;
            async_copy16((unsigned)(size_t)&As[row * LDA + q * 4],
                         A + (size_t)(m0 + row) * K + k0 + q * 4);
        }
#pragma unroll
        for (int i = 0; i < 2; ++i) {
            const int id = tid + 256 * i, row = id >> 3, q = id & 7;
            async_copy16((unsigned)(size_t)&Ws[row * LDA + q * 4],
                         W + (size_t)(n0 + row) * K + k0 + q * 4);
        }
        if (k0 + 32 < K)   // pull next K-chunk of A toward L2 (global_prefetch_b8)
            __builtin_prefetch(A + (size_t)(m0 + (tid >> 1)) * K + k0 + 32, 0, 1);

        wait_async();
        __syncthreads();

        // ---- A fragment: lane m = lmod, K pattern {0-7,16-23} / {8-15,24-31}
        v16h af;
#pragma unroll
        for (int j = 0; j < 16; ++j) {
            const int kk = (j >> 3) * 16 + (j & 7) + lhalf * 8;
            af[j] = (_Float16)As[(wave * 16 + lmod) * LDA + kk];
        }
        // ---- 4 B fragments (B[k][n] = W[n][k]): lane n = lmod, K pattern {0-7,16-23} / {8-15,24-31}
#pragma unroll
        for (int nt = 0; nt < 4; ++nt) {
            v16h bf;
#pragma unroll
            for (int j = 0; j < 16; ++j) {
                const int kk = (j >> 3) * 16 + (j & 7) + lhalf * 8;
                bf[j] = (_Float16)Ws[(nt * 16 + lmod) * LDA + kk];
            }
            acc[nt] = WMMA_F16(af, bf, acc[nt]);
        }
        __syncthreads();
    }

    // ---- epilogue: C/D layout -> lane holds (m = r + 8*lhalf, n = lmod)
#pragma unroll
    for (int nt = 0; nt < 4; ++nt) {
        const int n = n0 + nt * 16 + lmod;
        const float bv = bias ? bias[n] : 0.0f;
#pragma unroll
        for (int r = 0; r < 8; ++r) {
            const int m = m0 + wave * 16 + r + lhalf * 8;
            float v = acc[nt][r] + bv;
            if constexpr (FLAGS & FLAG_RELU) v = fmaxf(v, 0.0f);
            if constexpr (FLAGS & FLAG_PE) {
                // h = sqrt(H)*gemm + PE(s, n)
                const int   s     = m & (SEQ - 1);
                const float cpair = (float)(n & ~1);
                const float freq  = __expf(cpair * (-9.210340371976184f / 256.0f));
                const float ang   = (float)s * freq;
                const float pe    = (n & 1) ? __cosf(ang) : __sinf(ang);
                v = 16.0f * v + pe;
            }
            if constexpr (FLAGS & FLAG_RESIDUAL) v += residual[(size_t)m * N + n];
            out[(size_t)m * N + n] = v;
        }
    }
}

// ---------------------------------------------------------------------------
// LayerNorm over H=256, one wave per row, b128 loads/stores, wave32 shfl reduce.
// ---------------------------------------------------------------------------
__global__ __launch_bounds__(256)
void layernorm_rows(const float* __restrict__ x, const float* __restrict__ g,
                    const float* __restrict__ b, float* __restrict__ out)
{
    const int wave = threadIdx.x >> 5;
    const int lane = threadIdx.x & 31;
    const int row  = blockIdx.x * 8 + wave;

    const float4* xr = (const float4*)(x + (size_t)row * HDIM);
    const float4 a0 = xr[lane * 2];
    const float4 a1 = xr[lane * 2 + 1];

    float sum = a0.x + a0.y + a0.z + a0.w + a1.x + a1.y + a1.z + a1.w;
#pragma unroll
    for (int off = 16; off >= 1; off >>= 1) sum += __shfl_xor(sum, off, 32);
    const float mu = sum * (1.0f / HDIM);

    float var = 0.0f;
    {
        const float d0 = a0.x - mu, d1 = a0.y - mu, d2 = a0.z - mu, d3 = a0.w - mu;
        const float d4 = a1.x - mu, d5 = a1.y - mu, d6 = a1.z - mu, d7 = a1.w - mu;
        var = d0*d0 + d1*d1 + d2*d2 + d3*d3 + d4*d4 + d5*d5 + d6*d6 + d7*d7;
    }
#pragma unroll
    for (int off = 16; off >= 1; off >>= 1) var += __shfl_xor(var, off, 32);
    const float rstd = rsqrtf(var * (1.0f / HDIM) + 1e-5f);

    const float4* g4 = (const float4*)g;
    const float4* b4 = (const float4*)b;
    const float4 g0 = g4[lane * 2], g1 = g4[lane * 2 + 1];
    const float4 c0 = b4[lane * 2], c1 = b4[lane * 2 + 1];

    float4 o0, o1;
    o0.x = (a0.x - mu) * rstd * g0.x + c0.x;  o0.y = (a0.y - mu) * rstd * g0.y + c0.y;
    o0.z = (a0.z - mu) * rstd * g0.z + c0.z;  o0.w = (a0.w - mu) * rstd * g0.w + c0.w;
    o1.x = (a1.x - mu) * rstd * g1.x + c1.x;  o1.y = (a1.y - mu) * rstd * g1.y + c1.y;
    o1.z = (a1.z - mu) * rstd * g1.z + c1.z;  o1.w = (a1.w - mu) * rstd * g1.w + c1.w;

    float4* orow = (float4*)(out + (size_t)row * HDIM);
    orow[lane * 2]     = o0;
    orow[lane * 2 + 1] = o1;
}

// ---------------------------------------------------------------------------
// Sliding-window MHA. Block = (q-tile of 16, batch), 4 waves = 4 heads.
// Window |j - i| <= 32  =>  key tiles qt-2 .. qt+2 (80 real cols, padded to 96).
// Q*K^T: 2 chained WMMAs (K=64). P*V: 3 K=32 chunks x 4 dh-groups of WMMAs.
// All loops fully unrolled; WMMA never sits under divergent control flow.
// ---------------------------------------------------------------------------
__global__ __launch_bounds__(128)
void attn_window(const float* __restrict__ qkv, float* __restrict__ outp)
{
    __shared__ float sc[4][16][96];           // per-head score/prob tile (24 KB)

    const int head  = threadIdx.x >> 5;
    const int lane  = threadIdx.x & 31;
    const int lhalf = lane >> 4;
    const int lmod  = lane & 15;
    const int qt    = blockIdx.x;
    const int bb    = blockIdx.y;
    const int t0    = qt * 16;

    const float* base = qkv + (size_t)bb * SEQ * 768;
    float* scf = &sc[head][0][0];

    // init this head's tile to -inf (cols 80..95 stay -inf => P == 0 there)
    for (int i = lane; i < 16 * 96; i += 32) scf[i] = -__builtin_inff();

    // ---- Q fragments (A layout), softmax scale 1/sqrt(64) folded into Q
    v16h qa0, qa1;
    {
        const float* qrow = base + (size_t)(t0 + lmod) * 768 + head * 64;
#pragma unroll
        for (int j = 0; j < 16; ++j) {
            const int kk = (j >> 3) * 16 + (j & 7) + lhalf * 8;
            qa0[j] = (_Float16)(qrow[kk]      * 0.125f);
            qa1[j] = (_Float16)(qrow[32 + kk] * 0.125f);
        }
    }

    // ---- scores for the 5 key tiles in the band (uniform guards, unrolled)
#pragma unroll
    for (int dt = 0; dt < 5; ++dt) {
        const int kt = qt - 2 + dt;
        if (kt < 0 || kt >= SEQ / 16) continue;

        const float* krow = base + (size_t)(kt * 16 + lmod) * 768 + 256 + head * 64;
        v16h kb0, kb1;                       // B[k][n] = K[key n][dim k]
#pragma unroll
        for (int j = 0; j < 16; ++j) {
            const int kk = (j >> 3) * 16 + (j & 7) + lhalf * 8;
            kb0[j] = (_Float16)krow[kk];
            kb1[j] = (_Float16)krow[32 + kk];
        }
        v8f s = {};
        s = WMMA_F16(qa0, kb0, s);
        s = WMMA_F16(qa1, kb1, s);

        const int kj  = kt * 16 + lmod;      // global key index (lane's column)
        const int col = dt * 16 + lmod;      // LDS column (key - (t0-32))
#pragma unroll
        for (int r = 0; r < 8; ++r) {
            const int qi = t0 + r + lhalf * 8;
            const int dj = kj - qi;
            sc[head][r + lhalf * 8][col] =
                (dj <= 32 && dj >= -32) ? s[r] : -__builtin_inff();
        }
    }

    // ---- softmax, one row at a time across the full wave (3 cols/lane)
    for (int r = 0; r < 16; ++r) {
        float v0 = sc[head][r][lane];
        float v1 = sc[head][r][lane + 32];
        float v2 = sc[head][r][lane + 64];
        float mx = fmaxf(v0, fmaxf(v1, v2));
#pragma unroll
        for (int off = 16; off >= 1; off >>= 1) mx = fmaxf(mx, __shfl_xor(mx, off, 32));
        const float e0 = __expf(v0 - mx);
        const float e1 = __expf(v1 - mx);
        const float e2 = __expf(v2 - mx);
        float sum = e0 + e1 + e2;
#pragma unroll
        for (int off = 16; off >= 1; off >>= 1) sum += __shfl_xor(sum, off, 32);
        const float inv = 1.0f / sum;
        sc[head][r][lane]      = e0 * inv;
        sc[head][r][lane + 32] = e1 * inv;
        sc[head][r][lane + 64] = e2 * inv;
    }

    // ---- O = P @ V  (keys in 3 chunks of 32; dh=64 as 4 N-tiles of 16)
    v8f oacc[4] = {{}, {}, {}, {}};
#pragma unroll
    for (int c = 0; c < 3; ++c) {
        v16h pa;                              // A layout from LDS probabilities
#pragma unroll
        for (int j = 0; j < 16; ++j) {
            const int col = c * 32 + (j >> 3) * 16 + (j & 7) + lhalf * 8;
            pa[j] = (_Float16)sc[head][lmod][col];
        }
#pragma unroll
        for (int g = 0; g < 4; ++g) {
            v16h vb;                          // B[k][n] = V[key k][dim n]
#pragma unroll
            for (int j = 0; j < 16; ++j) {
                const int kloc = (j >> 3) * 16 + (j & 7) + lhalf * 8;
                const int key  = t0 - 32 + c * 32 + kloc;
                const float vv = (key >= 0 && key < SEQ)
                    ? base[(size_t)key * 768 + 512 + head * 64 + g * 16 + lmod]
                    : 0.0f;                   // P==0 there anyway
                vb[j] = (_Float16)vv;
            }
            oacc[g] = WMMA_F16(pa, vb, oacc[g]);
        }
    }

    // ---- store O, heads concatenated along features
#pragma unroll
    for (int g = 0; g < 4; ++g) {
        const int colo = head * 64 + g * 16 + lmod;
#pragma unroll
        for (int r = 0; r < 8; ++r) {
            const int tok = bb * SEQ + t0 + r + lhalf * 8;
            outp[(size_t)tok * HDIM + colo] = oacc[g][r];
        }
    }
}

// ---------------------------------------------------------------------------
// Host-side orchestration. Workspace layout (floats), total ~80 MB:
//   h[8192*256] | tmp[8192*256] | qkv[8192*768] | attn[8192*256] | mid[8192*1024]
// ---------------------------------------------------------------------------
extern "C" void kernel_launch(void* const* d_in, const int* in_sizes, int n_in,
                              void* d_out, int out_size, void* d_ws, size_t ws_size,
                              hipStream_t stream)
{
    const float* x     = (const float*)d_in[0];
    const float* w_in  = (const float*)d_in[1];
    const float* w_out = (const float*)d_in[2];
    const float* b_out = (const float*)d_in[3];
    const float* qkv_w = (const float*)d_in[4];
    const float* qkv_b = (const float*)d_in[5];
    const float* out_w = (const float*)d_in[6];
    const float* out_b = (const float*)d_in[7];
    const float* ln1_g = (const float*)d_in[8];
    const float* ln1_b = (const float*)d_in[9];
    const float* ln2_g = (const float*)d_in[10];
    const float* ln2_b = (const float*)d_in[11];
    const float* ff1_w = (const float*)d_in[12];
    const float* ff1_b = (const float*)d_in[13];
    const float* ff2_w = (const float*)d_in[14];
    const float* ff2_b = (const float*)d_in[15];
    const float* lnf_g = (const float*)d_in[16];
    const float* lnf_b = (const float*)d_in[17];

    float* h    = (float*)d_ws;
    float* tmp  = h    + (size_t)TOKENS * 256;
    float* qkvb = tmp  + (size_t)TOKENS * 256;
    float* attn = qkvb + (size_t)TOKENS * 768;
    float* mid  = attn + (size_t)TOKENS * 256;

    const int M = TOKENS;
    const dim3 blk(256);
    const dim3 ablk(128);

    // encode: h = 16*(x @ w_in^T) + PE
    gemm_wmma<FLAG_PE><<<dim3(M / 128, 256 / 64), blk, 0, stream>>>(
        x, w_in, nullptr, nullptr, h, M, 256, 256);

    for (int l = 0; l < 3; ++l) {
        layernorm_rows<<<M / 8, blk, 0, stream>>>(h, ln1_g + l * 256, ln1_b + l * 256, tmp);

        gemm_wmma<0><<<dim3(M / 128, 768 / 64), blk, 0, stream>>>(
            tmp, qkv_w + (size_t)l * 768 * 256, qkv_b + l * 768, nullptr,
            qkvb, M, 768, 256);

        attn_window<<<dim3(SEQ / 16, 4), ablk, 0, stream>>>(qkvb, attn);

        gemm_wmma<FLAG_RESIDUAL><<<dim3(M / 128, 256 / 64), blk, 0, stream>>>(
            attn, out_w + (size_t)l * 256 * 256, out_b + l * 256, h,
            h, M, 256, 256);

        layernorm_rows<<<M / 8, blk, 0, stream>>>(h, ln2_g + l * 256, ln2_b + l * 256, tmp);

        gemm_wmma<FLAG_RELU><<<dim3(M / 128, 1024 / 64), blk, 0, stream>>>(
            tmp, ff1_w + (size_t)l * 1024 * 256, ff1_b + l * 1024, nullptr,
            mid, M, 1024, 256);

        gemm_wmma<FLAG_RESIDUAL><<<dim3(M / 128, 256 / 64), blk, 0, stream>>>(
            mid, ff2_w + (size_t)l * 256 * 1024, ff2_b + l * 256, h,
            h, M, 256, 1024);
    }

    layernorm_rows<<<M / 8, blk, 0, stream>>>(h, lnf_g, lnf_b, tmp);

    gemm_wmma<0><<<dim3(M / 128, 256 / 64), blk, 0, stream>>>(
        tmp, w_out, b_out, nullptr, (float*)d_out, M, 256, 256);
}
